// GCN_10153302688022
// MI455X (gfx1250) — matrix operand order repaired and checked by reading, not compiled, not run on hardware
//
#include <hip/hip_runtime.h>

// ---------------------------------------------------------------------------
// 2-layer GAT (H=4, C=64) for MI455X / gfx1250 (wave32, WMMA, async-to-LDS).
// GEMMs: inputs pre-converted to f16, tiles double-buffered in LDS via
// global_load_async_to_lds_b128 (ASYNCcnt pipeline), MACs on
// v_wmma_f32_16x16x32_f16. Edge softmax/aggregation via L2-resident atomics
// (feature matrices are 51.2MB each -> resident in the 192MB L2).
// ---------------------------------------------------------------------------

#define NN_NODES 50000
#define N_IN     128
#define HCHAN    256   // H * C = 4 * 64
#define NHEAD    4
#define CHEAD    64

typedef _Float16 v8h  __attribute__((ext_vector_type(8)));
typedef _Float16 v16h __attribute__((ext_vector_type(16)));
typedef float    v8f  __attribute__((ext_vector_type(8)));

union V16H { v16h v; v8h h[2]; };

// Async 16B copy: global -> LDS, tracked by ASYNCcnt (CDNA5 §15.18.3 op 98).
__device__ __forceinline__ void async_load_b128(unsigned lds_off, const void* gaddr) {
  asm volatile("global_load_async_to_lds_b128 %0, %1, off"
               :: "v"(lds_off), "v"(gaddr)
               : "memory");
}

// ---------------------------------------------------------------------------
// WMMA GEMM: Y[M x 256] = A[M x K] * Wt[256 x K]^T   (A, Wt are f16; K = 128/256)
// Block tile 128x64, 8 waves (4x2), each wave 32x32 = 2x2 WMMA 16x16x32 tiles.
// Double-buffered async LDS staging: tile t+1 streams in while tile t computes.
// ---------------------------------------------------------------------------
#define TILE_M 128
#define TILE_N 64
#define LDSP   40   // f16 pitch (80B rows; keeps 16B alignment for chunks)

__global__ __launch_bounds__(256) void gat_gemm_wmma(
    const _Float16* __restrict__ A, const _Float16* __restrict__ Wt,
    float* __restrict__ Y, int M, int K)
{
  __shared__ __align__(16) _Float16 As[2][TILE_M * LDSP];
  __shared__ __align__(16) _Float16 Bs[2][TILE_N * LDSP];

  const int tid  = threadIdx.x;
  const int lane = tid & 31;
  const int wave = tid >> 5;
  const int wm   = wave >> 1;      // 0..3
  const int wn   = wave & 1;       // 0..1
  const int half = lane >> 4;      // 0..1
  const int l16  = lane & 15;
  const int blockM = blockIdx.x * TILE_M;
  const int blockN = blockIdx.y * TILE_N;

  // LDS byte offsets (low 32 bits of shared-aperture flat address).
  const unsigned asBase0 = (unsigned)(size_t)&As[0][0];
  const unsigned asBase1 = (unsigned)(size_t)&As[1][0];
  const unsigned bsBase0 = (unsigned)(size_t)&Bs[0][0];
  const unsigned bsBase1 = (unsigned)(size_t)&Bs[1][0];

  // Per-thread async-copy geometry: 16B chunks, 4 chunks per 32-f16 row.
  const int tRow = tid >> 2;   // 0..63
  const int tChk = tid & 3;    // 0..3

  // Issue the 3 async b128 copies (per wave) for k-tile t into buffer `buf`.
  auto issue_tile = [&](int t, int buf) {
    const int k0 = t << 5;
    const unsigned asB = buf ? asBase1 : asBase0;
    const unsigned bsB = buf ? bsBase1 : bsBase0;
    #pragma unroll
    for (int i = 0; i < 2; ++i) {                 // A: 128 rows x 64B
      const int row = tRow + i * 64;
      int gr = blockM + row; gr = (gr < M) ? gr : (M - 1);
      async_load_b128(asB + row * (LDSP * 2) + tChk * 16,
                      A + (size_t)gr * K + k0 + tChk * 8);
    }
    async_load_b128(bsB + tRow * (LDSP * 2) + tChk * 16,  // B: 64 rows x 64B
                    Wt + (size_t)(blockN + tRow) * K + k0 + tChk * 8);
  };

  v8f acc[2][2] = {};
  const int T = K >> 5;        // K/32 k-tiles

  issue_tile(0, 0);            // prologue: ASYNCcnt = 3
  for (int t = 0; t < T; ++t) {
    const int cb = t & 1;
    if (t + 1 < T) {
      issue_tile(t + 1, cb ^ 1);                          // ASYNCcnt -> 6
      asm volatile("s_wait_asynccnt 0x3" ::: "memory");   // tile t complete (in-order)
    } else {
      asm volatile("s_wait_asynccnt 0x0" ::: "memory");
    }
    __syncthreads();           // tile t visible from all waves

    const _Float16* Ab = &As[cb][0];
    const _Float16* Bb = &Bs[cb][0];
    V16H a[2], b[2];
    #pragma unroll
    for (int q = 0; q < 2; ++q) {
      // A fragment: row = l16, k chunks at half*8 and 16+half*8 (ISA 16-bit A layout)
      const int m = wm * 32 + q * 16 + l16;
      a[q].h[0] = *(const v8h*)&Ab[m * LDSP + half * 8];
      a[q].h[1] = *(const v8h*)&Ab[m * LDSP + 16 + half * 8];
      // B fragment: col = l16, k chunks at half*8 and 16+half*8 (same as A)
      const int n = wn * 32 + q * 16 + l16;
      b[q].h[0] = *(const v8h*)&Bb[n * LDSP + half * 8];
      b[q].h[1] = *(const v8h*)&Bb[n * LDSP + 16 + half * 8];
    }
    #pragma unroll
    for (int tm = 0; tm < 2; ++tm)
      #pragma unroll
      for (int tn = 0; tn < 2; ++tn)
        acc[tm][tn] = __builtin_amdgcn_wmma_f32_16x16x32_f16(
            false, a[tm].v, false, b[tn].v, (short)0, acc[tm][tn], false, false);
    __syncthreads();           // all waves done reading buffer cb before re-issue
  }

  // Store: C/D layout -> m = r + half*8, n = l16. Uniform fast path for full tiles.
  const bool full = (blockM + TILE_M) <= M;
  #pragma unroll
  for (int tm = 0; tm < 2; ++tm) {
    #pragma unroll
    for (int tn = 0; tn < 2; ++tn) {
      const int nCol = blockN + wn * 32 + tn * 16 + l16;
      const int gr0  = blockM + wm * 32 + tm * 16 + half * 8;
      float* yp = Y + (size_t)gr0 * HCHAN + nCol;
      if (full) {
        #pragma unroll
        for (int r = 0; r < 8; ++r) yp[(size_t)r * HCHAN] = acc[tm][tn][r];
      } else {
        #pragma unroll
        for (int r = 0; r < 8; ++r)
          if (gr0 + r < M) yp[(size_t)r * HCHAN] = acc[tm][tn][r];
      }
    }
  }
}

// ---------------------------------------------------------------------------
// f32 -> f16 conversion / weight transpose (run once per call, tiny cost)
// ---------------------------------------------------------------------------
__global__ __launch_bounds__(256) void gat_cvt_f16(
    const float* __restrict__ in, _Float16* __restrict__ out, long long n)
{
  const long long i = (long long)blockIdx.x * blockDim.x + threadIdx.x;
  if (i < n) out[i] = (_Float16)in[i];
}

// Wt[n][k] = W[k][n] as f16 (W is K x 256 row-major)
__global__ __launch_bounds__(256) void gat_wt_f16(
    const float* __restrict__ W, _Float16* __restrict__ Wt, int K)
{
  const int i = blockIdx.x * blockDim.x + threadIdx.x;
  if (i >= HCHAN * K) return;
  const int n = i / K, k = i - n * K;
  Wt[(size_t)n * K + k] = (_Float16)W[(size_t)k * HCHAN + n];
}

// ---------------------------------------------------------------------------
// alpha_s / alpha_d: per (node, head) dot of h[n,head,:] with attention vectors
// ---------------------------------------------------------------------------
__global__ __launch_bounds__(256) void gat_alpha(
    const float* __restrict__ h, const float* __restrict__ asrc,
    const float* __restrict__ adst, float* __restrict__ als,
    float* __restrict__ ald, int Nn)
{
  const int idx = blockIdx.x * blockDim.x + threadIdx.x;
  if (idx >= Nn * NHEAD) return;
  const int node = idx >> 2, head = idx & 3;
  const float4* hp = (const float4*)(h + (size_t)node * HCHAN + head * CHEAD);
  const float4* ap = (const float4*)(asrc + head * CHEAD);
  const float4* bp = (const float4*)(adst + head * CHEAD);
  float ss = 0.f, sd = 0.f;
  #pragma unroll
  for (int i = 0; i < 16; ++i) {
    const float4 v = hp[i], a = ap[i], b = bp[i];
    ss += v.x * a.x + v.y * a.y + v.z * a.z + v.w * a.w;
    sd += v.x * b.x + v.y * b.y + v.z * b.z + v.w * b.w;
  }
  als[idx] = ss;
  ald[idx] = sd;
}

// Order-preserving float<->uint encoding for atomic max
__device__ __forceinline__ unsigned ford(float f) {
  const unsigned u = __float_as_uint(f);
  return (u & 0x80000000u) ? ~u : (u | 0x80000000u);
}
__device__ __forceinline__ float forddec(unsigned u) {
  return (u & 0x80000000u) ? __uint_as_float(u & 0x7FFFFFFFu)
                           : __uint_as_float(~u);
}
__device__ __forceinline__ float lrelu(float x) { return x > 0.f ? x : 0.2f * x; }

// ---------------------------------------------------------------------------
// Edge pass 1: segment max of leaky_relu(alpha_s[src] + alpha_d[dst]) per head.
// Edges e >= E are the implicit self-loops (src = dst = e - E).
// ---------------------------------------------------------------------------
__global__ __launch_bounds__(256) void gat_edge_max(
    const int* __restrict__ src, const int* __restrict__ dst,
    const float* __restrict__ als, const float* __restrict__ ald,
    unsigned* __restrict__ emax, int E, int Etot)
{
  const int e = blockIdx.x * blockDim.x + threadIdx.x;
  if (e >= Etot) return;
  int s, d;
  if (e < E) { s = src[e]; d = dst[e]; } else { s = e - E; d = s; }
  const float4 a = *(const float4*)(als + (size_t)s * 4);
  const float4 b = *(const float4*)(ald + (size_t)d * 4);
  const float v[4] = { lrelu(a.x + b.x), lrelu(a.y + b.y),
                       lrelu(a.z + b.z), lrelu(a.w + b.w) };
  #pragma unroll
  for (int h = 0; h < 4; ++h) atomicMax(&emax[(size_t)d * 4 + h], ford(v[h]));
}

// ---------------------------------------------------------------------------
// Edge pass 2: p = exp(e - max), denom[dst] += p; p stored for the aggregate pass
// ---------------------------------------------------------------------------
__global__ __launch_bounds__(256) void gat_edge_exp(
    const int* __restrict__ src, const int* __restrict__ dst,
    const float* __restrict__ als, const float* __restrict__ ald,
    const unsigned* __restrict__ emax, float* __restrict__ p,
    float* __restrict__ denom, int E, int Etot)
{
  const int e = blockIdx.x * blockDim.x + threadIdx.x;
  if (e >= Etot) return;
  int s, d;
  if (e < E) { s = src[e]; d = dst[e]; } else { s = e - E; d = s; }
  const float4 a = *(const float4*)(als + (size_t)s * 4);
  const float4 b = *(const float4*)(ald + (size_t)d * 4);
  const float v[4] = { lrelu(a.x + b.x), lrelu(a.y + b.y),
                       lrelu(a.z + b.z), lrelu(a.w + b.w) };
  float pv[4];
  #pragma unroll
  for (int h = 0; h < 4; ++h) {
    const float m = forddec(emax[(size_t)d * 4 + h]);
    pv[h] = __expf(v[h] - m);
    atomicAdd(&denom[(size_t)d * 4 + h], pv[h]);
  }
  *(float4*)(p + (size_t)e * 4) = make_float4(pv[0], pv[1], pv[2], pv[3]);
}

// ---------------------------------------------------------------------------
// Edge pass 3: out[dst] += h[src] * (p / denom[dst]); 64 threads per edge,
// each thread covers one float4 of the 256-channel row. L2-resident atomics.
// ---------------------------------------------------------------------------
__global__ __launch_bounds__(256) void gat_aggregate(
    const int* __restrict__ src, const int* __restrict__ dst,
    const float* __restrict__ h, const float* __restrict__ p,
    const float* __restrict__ denom, float* __restrict__ out, int E, int Etot)
{
  const long long t = (long long)blockIdx.x * blockDim.x + threadIdx.x;
  const long long e = t >> 6;
  if (e >= (long long)Etot) return;
  const int c = ((int)t & 63) * 4;   // 0..252
  const int head = c >> 6;           // 0..3
  int s, d;
  if (e < E) { s = src[e]; d = dst[e]; } else { s = (int)(e - E); d = s; }
  const float alpha = p[e * 4 + head] / (denom[(size_t)d * 4 + head] + 1e-16f);
  const float4 hv = *(const float4*)(h + (size_t)s * HCHAN + c);
  float* o = out + (size_t)d * HCHAN + c;
  atomicAdd(o + 0, hv.x * alpha);
  atomicAdd(o + 1, hv.y * alpha);
  atomicAdd(o + 2, hv.z * alpha);
  atomicAdd(o + 3, hv.w * alpha);
}

// ---------------------------------------------------------------------------
// Finalizers
// ---------------------------------------------------------------------------
// Layer-1 output: relu(acc + b1) written directly as f16 (feeds GEMM2 only).
__global__ __launch_bounds__(256) void gat_relu_bias_f16(
    const float* __restrict__ acc, const float* __restrict__ b,
    _Float16* __restrict__ o, int total)
{
  const int i = blockIdx.x * blockDim.x + threadIdx.x;
  if (i >= total) return;
  const float v = acc[i] + b[i & (HCHAN - 1)];
  o[i] = (_Float16)(v > 0.f ? v : 0.f);
}

__global__ __launch_bounds__(256) void gat_mean_bias(
    const float* __restrict__ acc, const float* __restrict__ b,
    float* __restrict__ o, int Nn)
{
  const int i = blockIdx.x * blockDim.x + threadIdx.x;
  if (i >= Nn * CHEAD) return;
  const int n = i >> 6, c = i & 63;
  const float* a = acc + (size_t)n * HCHAN + c;
  o[i] = 0.25f * (a[0] + a[64] + a[128] + a[192]) + b[c];
}

__global__ __launch_bounds__(256) void fill_u32(
    unsigned* __restrict__ p, unsigned v, long long n)
{
  const long long i = (long long)blockIdx.x * blockDim.x + threadIdx.x;
  if (i < n) p[i] = v;
}

// ---------------------------------------------------------------------------
// Host orchestration
// ---------------------------------------------------------------------------
extern "C" void kernel_launch(void* const* d_in, const int* in_sizes, int n_in,
                              void* d_out, int out_size, void* d_ws, size_t ws_size,
                              hipStream_t stream)
{
  const int NN = NN_NODES;
  const float* x   = (const float*)d_in[0];
  const int*   ei  = (const int*)d_in[1];     // edge_index [2, E]
  const float* W1  = (const float*)d_in[2];
  const float* as1 = (const float*)d_in[3];
  const float* ad1 = (const float*)d_in[4];
  const float* b1  = (const float*)d_in[5];
  const float* W2  = (const float*)d_in[6];
  const float* as2 = (const float*)d_in[7];
  const float* ad2 = (const float*)d_in[8];
  const float* b2  = (const float*)d_in[9];
  float* out = (float*)d_out;

  const int E    = in_sizes[1] / 2;
  const int Etot = E + NN;
  const int* src = ei;
  const int* dst = ei + E;

  // Workspace layout (all chunks 16B aligned):
  char* ws = (char*)d_ws;
  const size_t featBytes = (size_t)NN * HCHAN * sizeof(float);   // 51.2 MB
  float*    bufA  = (float*)ws;                                  // h1 -> out2
  float*    bufB  = (float*)(ws + featBytes);                    // out1 -> h2
  char*     cur   = ws + 2 * featBytes;
  float*    pbuf  = (float*)cur;              cur += (size_t)Etot * 4 * sizeof(float);
  float*    als   = (float*)cur;              cur += (size_t)NN * 4 * sizeof(float);
  float*    ald   = (float*)cur;              cur += (size_t)NN * 4 * sizeof(float);
  unsigned* emax  = (unsigned*)cur;           cur += (size_t)NN * 4 * sizeof(unsigned);
  float*    denom = (float*)cur;              cur += (size_t)NN * 4 * sizeof(float);
  _Float16* xh    = (_Float16*)cur;           cur += (size_t)NN * N_IN * 2;    // 12.8 MB
  _Float16* hrh   = (_Float16*)cur;           cur += (size_t)NN * HCHAN * 2;   // 25.6 MB
  _Float16* wt1   = (_Float16*)cur;           cur += (size_t)HCHAN * N_IN * 2;
  _Float16* wt2   = (_Float16*)cur;           cur += (size_t)HCHAN * HCHAN * 2;

  const int TB = 256;
  const long long featN = (long long)NN * HCHAN;
  const long long xN    = (long long)NN * N_IN;
  const long long aggT  = (long long)Etot * 64;
  dim3 gGemm((NN + TILE_M - 1) / TILE_M, HCHAN / TILE_N);
  const int gEdge  = (Etot + TB - 1) / TB;
  const int gAlpha = (NN * NHEAD + TB - 1) / TB;
  const int gFeat  = (int)((featN + TB - 1) / TB);
  const int gAgg   = (int)((aggT + TB - 1) / TB);

  // -------- Precompute f16 operands --------
  gat_cvt_f16<<<(int)((xN + TB - 1) / TB), TB, 0, stream>>>(x, xh, xN);
  gat_wt_f16<<<(HCHAN * N_IN + TB - 1) / TB, TB, 0, stream>>>(W1, wt1, N_IN);
  gat_wt_f16<<<(HCHAN * HCHAN + TB - 1) / TB, TB, 0, stream>>>(W2, wt2, HCHAN);

  // ---------------- Layer 1 ----------------
  fill_u32<<<gFeat, TB, 0, stream>>>((unsigned*)bufB, 0u, featN);         // out1 = 0
  fill_u32<<<(NN * 8 + TB - 1) / TB, TB, 0, stream>>>(emax, 0u, NN * 8);  // emax(-inf enc) + denom
  gat_gemm_wmma<<<gGemm, TB, 0, stream>>>(xh, wt1, bufA, NN, N_IN);       // h1 = x @ W1
  gat_alpha<<<gAlpha, TB, 0, stream>>>(bufA, as1, ad1, als, ald, NN);
  gat_edge_max<<<gEdge, TB, 0, stream>>>(src, dst, als, ald, emax, E, Etot);
  gat_edge_exp<<<gEdge, TB, 0, stream>>>(src, dst, als, ald, emax, pbuf, denom, E, Etot);
  gat_aggregate<<<gAgg, TB, 0, stream>>>(src, dst, bufA, pbuf, denom, bufB, E, Etot);
  gat_relu_bias_f16<<<gFeat, TB, 0, stream>>>(bufB, b1, hrh, (int)featN); // hrelu (f16)

  // ---------------- Layer 2 ----------------
  gat_gemm_wmma<<<gGemm, TB, 0, stream>>>(hrh, wt2, bufB, NN, HCHAN);     // h2 = hrelu @ W2
  gat_alpha<<<gAlpha, TB, 0, stream>>>(bufB, as2, ad2, als, ald, NN);
  fill_u32<<<gFeat, TB, 0, stream>>>((unsigned*)bufA, 0u, featN);         // out2 = 0
  fill_u32<<<(NN * 8 + TB - 1) / TB, TB, 0, stream>>>(emax, 0u, NN * 8);
  gat_edge_max<<<gEdge, TB, 0, stream>>>(src, dst, als, ald, emax, E, Etot);
  gat_edge_exp<<<gEdge, TB, 0, stream>>>(src, dst, als, ald, emax, pbuf, denom, E, Etot);
  gat_aggregate<<<gAgg, TB, 0, stream>>>(src, dst, bufB, pbuf, denom, bufA, E, Etot);
  gat_mean_bias<<<(NN * CHEAD + TB - 1) / TB, TB, 0, stream>>>(bufA, b2, out, NN);
}
